// HMM_4398046511418
// MI455X (gfx1250) — matrix operand order repaired and checked by reading, not compiled, not run on hardware
//
#include <hip/hip_runtime.h>
#include <math.h>

// ============================================================================
// HMM forward on MI455X (gfx1250), restructured as a chunked matrix-product
// scan so the ~4.4 TFLOP of folding runs on WMMA bf16 matrix cores instead of
// a serial 2047-step vec-mat recurrence.
//
//   out = (alpha0 @ A) * prod_{t=1..2046} (diag(e_t) @ A) * e_2047
//
// ws layout (requires ~144 MB):
//   [0,2MB)      A        : softmax(q_transition) rows, bf16 [1024x1024]
//   [2,10MB)     E        : emission probs per step, f32 [2048x1024]
//   [10,14MB)    Ebf      : same, bf16 (for packed row-scaling in the GEMM)
//   [14MB..)     p0/mx/inv: small f32 vectors
//   [16,80MB)    P ping   : 32 chunk products, bf16 [32][1024][1024]
//   [80,144MB)   P pong
// ============================================================================

typedef __attribute__((ext_vector_type(16))) __bf16 v16bf;
typedef __attribute__((ext_vector_type(8)))  __bf16 v8bf;
typedef __attribute__((ext_vector_type(8)))  float  v8f;

#define S_DIM   1024
#define T_LEN   2048
#define V_OBS   32000
#define NCHUNK  32
#define CLEN    64        // steps per chunk (last chunk short)
#define CH_ELEM ((size_t)1 << 20)   // 1024*1024 elements per chunk product

static constexpr size_t A_OFF   = 0;
static constexpr size_t E_OFF   = (size_t)2  << 20;
static constexpr size_t EBF_OFF = (size_t)10 << 20;
static constexpr size_t P0V_OFF = (size_t)14 << 20;
static constexpr size_t MX_OFF  = P0V_OFF + 4096;
static constexpr size_t INV_OFF = P0V_OFF + 8192;
static constexpr size_t PB0_OFF = (size_t)16 << 20;
static constexpr size_t PB1_OFF = (size_t)80 << 20;

// ---------------- block reductions (256 threads) ----------------------------
__device__ __forceinline__ float blkMax256(float v, float* sm) {
    int tid = threadIdx.x;
    __syncthreads();
    sm[tid] = v; __syncthreads();
    for (int s = 128; s >= 1; s >>= 1) {
        if (tid < s) sm[tid] = fmaxf(sm[tid], sm[tid + s]);
        __syncthreads();
    }
    return sm[0];
}
__device__ __forceinline__ float blkSum256(float v, float* sm) {
    int tid = threadIdx.x;
    __syncthreads();
    sm[tid] = v; __syncthreads();
    for (int s = 128; s >= 1; s >>= 1) {
        if (tid < s) sm[tid] += sm[tid + s];
        __syncthreads();
    }
    return sm[0];
}

// ---------------- softmax of q_initial -> p0 --------------------------------
__global__ void hmm_softmax_init(const float* __restrict__ q, float* __restrict__ p) {
    __shared__ float sm[256];
    int tid = threadIdx.x;
    float v[4], mx = -INFINITY;
    for (int i = 0; i < 4; ++i) { v[i] = q[tid + i * 256]; mx = fmaxf(mx, v[i]); }
    mx = blkMax256(mx, sm);
    float s = 0.f;
    for (int i = 0; i < 4; ++i) { v[i] = __expf(v[i] - mx); s += v[i]; }
    s = blkSum256(s, sm);
    float inv = 1.f / s;
    for (int i = 0; i < 4; ++i) p[tid + i * 256] = v[i] * inv;
}

// ---------------- row softmax of q_transition -> A (bf16) -------------------
__global__ void hmm_softmax_trans(const float* __restrict__ q, __bf16* __restrict__ A) {
    __shared__ float sm[256];
    int row = blockIdx.x, tid = threadIdx.x;
    const float* r = q + (size_t)row * S_DIM;
    float v[4], mx = -INFINITY;
    for (int i = 0; i < 4; ++i) { v[i] = r[tid + i * 256]; mx = fmaxf(mx, v[i]); }
    mx = blkMax256(mx, sm);
    float s = 0.f;
    for (int i = 0; i < 4; ++i) { v[i] = __expf(v[i] - mx); s += v[i]; }
    s = blkSum256(s, sm);
    float inv = 1.f / s;
    __bf16* o = A + (size_t)row * S_DIM;
    for (int i = 0; i < 4; ++i) o[tid + i * 256] = (__bf16)(v[i] * inv);
}

// ---------------- emission row stats: max + 1/sumexp over 32000 -------------
__global__ void hmm_emis_stats(const float* __restrict__ qe,
                               float* __restrict__ mxv, float* __restrict__ invv) {
    __shared__ float sm[256];
    int row = blockIdx.x, tid = threadIdx.x;
    const float* r = qe + (size_t)row * V_OBS;
    float mx = -INFINITY;
    for (int i = 0; i < V_OBS / 256; ++i) mx = fmaxf(mx, r[tid + i * 256]);
    mx = blkMax256(mx, sm);
    float s = 0.f;
    for (int i = 0; i < V_OBS / 256; ++i) s += __expf(r[tid + i * 256] - mx);
    s = blkSum256(s, sm);
    if (tid == 0) { mxv[row] = mx; invv[row] = 1.f / s; }
}

// ---------------- E[t][s] = softmax(q_e)[s, obs[t]]  (f32 + bf16 mirror) ----
__global__ void hmm_build_E(const float* __restrict__ qe, const int* __restrict__ obs,
                            const float* __restrict__ mxv, const float* __restrict__ invv,
                            float* __restrict__ E, __bf16* __restrict__ Ebf) {
    int t = blockIdx.x, tid = threadIdx.x;
    int o = obs[t];
    for (int i = 0; i < 4; ++i) {
        int s = tid + i * 256;
        float e = __expf(qe[(size_t)s * V_OBS + o] - mxv[s]) * invv[s];
        E[(size_t)t * S_DIM + s]   = e;
        Ebf[(size_t)t * S_DIM + s] = (__bf16)e;
    }
}

// ---------------- init chunk products: P[c] = diag(e_{t0}) @ A --------------
__global__ void hmm_initP(const __bf16* __restrict__ A, const float* __restrict__ E,
                          __bf16* __restrict__ P) {
    int k = blockIdx.x, c = blockIdx.y, tid = threadIdx.x;
    int t0 = 1 + c * CLEN;
    float ev = E[(size_t)t0 * S_DIM + k];
    const __bf16* a = A + (size_t)k * S_DIM;
    __bf16* p = P + (size_t)c * CH_ELEM + (size_t)k * S_DIM;
    for (int i = 0; i < 4; ++i) {
        int j = tid + i * 256;
        p[j] = (__bf16)(ev * (float)a[j]);
    }
}

// ---------------- WMMA GEMM: Pout[c] = Pa[c*aMul] @ (scale? diag(e_t):I) Pb --
// 256 threads / 8 waves; block = one 128x128 tile; wave = 64x32 (4x2 WMMA accs).
// SCALED=true  (steps): aMul=1, bMul=0, bAdd=0, Pb=A, t = 1 + c*CLEN + stepIdx;
//                       B rows scaled by e_t via packed bf16 multiply.
// SCALED=false (tree):  aMul=2, bMul=2, bAdd=1, Pb=Pa (pairwise combine).
template <bool SCALED>
__global__ void hmm_gemm(__bf16* __restrict__ Pout, const __bf16* __restrict__ Pa,
                         const __bf16* __restrict__ Pb,
                         int aMul, int bMul, int bAdd,
                         const __bf16* __restrict__ Ebf, int stepIdx) {
    const int chunk = blockIdx.y;
    const size_t aOff = (size_t)(chunk * aMul) * CH_ELEM;
    const size_t bOff = (size_t)(chunk * bMul + bAdd) * CH_ELEM;
    const size_t oOff = (size_t)chunk * CH_ELEM;
    const int tid  = threadIdx.x;
    const int row0 = (blockIdx.x >> 3) * 128;
    const int col0 = (blockIdx.x & 7) * 128;

    const __bf16* eRow = nullptr;
    if (SCALED) {
        int t = 1 + chunk * CLEN + stepIdx;
        if (t > T_LEN - 2) {
            // chunk already finished -> pass tile through unchanged
            for (int q = 0; q < 8; ++q) {
                int lin = tid + q * 256;
                int r = lin >> 4, cc = lin & 15;
                const uint4* s = reinterpret_cast<const uint4*>(
                    Pa + aOff + (size_t)(row0 + r) * S_DIM + col0) + cc;
                uint4* d = reinterpret_cast<uint4*>(
                    Pout + oOff + (size_t)(row0 + r) * S_DIM + col0) + cc;
                *d = *s;
            }
            return;
        }
        eRow = Ebf + (size_t)t * S_DIM;
    }

    const int wave = tid >> 5, lane = tid & 31;
    const int wr = row0 + (wave >> 2) * 64;   // 2 waves along rows
    const int wc = col0 + (wave & 3) * 32;    // 4 waves along cols
    const int m = lane & 15, half = lane >> 4;

    v8f acc[4][2] = {};

    for (int k0 = 0; k0 < S_DIM; k0 += 32) {
        // A fragments: 16x32 bf16, lane(m,half) holds K = half*8+{0..7} and 16+half*8+{0..7}
        v16bf afr[4];
#pragma unroll
        for (int rt = 0; rt < 4; ++rt) {
            const __bf16* p = Pa + aOff + (size_t)(wr + rt * 16 + m) * S_DIM + k0 + half * 8;
            v8bf lo = *reinterpret_cast<const v8bf*>(p);
            v8bf hi = *reinterpret_cast<const v8bf*>(p + 16);
            afr[rt] = __builtin_shufflevector(lo, hi, 0,1,2,3,4,5,6,7,8,9,10,11,12,13,14,15);
        }
        // B fragments: 32x16 bf16, lane(n,half) holds col n, K = half*8+{0..7} and 16+half*8+{0..7}
        // Row-scale by e_t[k] per element -> v_pk_mul_bf16.
        v16bf evv;
        if (SCALED) {
#pragma unroll
            for (int e2 = 0; e2 < 16; ++e2) evv[e2] = eRow[k0 + half * 8 + (e2 & 7) + (e2 >> 3) * 16];
        }
        v16bf bfr[2];
#pragma unroll
        for (int ct = 0; ct < 2; ++ct) {
            const __bf16* p = Pb + bOff + (size_t)(k0 + half * 8) * S_DIM + wc + ct * 16 + m;
            v16bf bb;
#pragma unroll
            for (int e2 = 0; e2 < 16; ++e2) bb[e2] = p[(size_t)((e2 & 7) + (e2 >> 3) * 16) * S_DIM];
            if (SCALED) bb = bb * evv;        // packed bf16 multiply
            bfr[ct] = bb;
        }
#pragma unroll
        for (int rt = 0; rt < 4; ++rt)
#pragma unroll
            for (int ct = 0; ct < 2; ++ct)
                acc[rt][ct] = __builtin_amdgcn_wmma_f32_16x16x32_bf16(
                    false, afr[rt], false, bfr[ct], (short)0, acc[rt][ct], false, false);
    }

    // D layout: VGPR r -> row half*8 + r, col = lane&15
#pragma unroll
    for (int rt = 0; rt < 4; ++rt)
#pragma unroll
        for (int ct = 0; ct < 2; ++ct)
#pragma unroll
            for (int r = 0; r < 8; ++r) {
                int row = wr + rt * 16 + half * 8 + r;
                int col = wc + ct * 16 + m;
                Pout[oOff + (size_t)row * S_DIM + col] = (__bf16)acc[rt][ct][r];
            }
}

// ---------------- final: out = ((p0*E0) @ A) @ G . E_{T-1} ------------------
__global__ void hmm_final(const float* __restrict__ p0, const float* __restrict__ E,
                          const __bf16* __restrict__ A, const __bf16* __restrict__ G,
                          float* __restrict__ out) {
    __shared__ float sa[S_DIM];
    __shared__ float sv[S_DIM];
    int j = threadIdx.x;
    sa[j] = p0[j] * E[j];                       // alpha0
    __syncthreads();
    float acc = 0.f;
    for (int k = 0; k < S_DIM; ++k)             // v0 = alpha0 @ A
        acc += sa[k] * (float)A[(size_t)k * S_DIM + j];
    __syncthreads();
    sv[j] = acc;
    __syncthreads();
    acc = 0.f;
    for (int k = 0; k < S_DIM; ++k)             // u = v0 @ G
        acc += sv[k] * (float)G[(size_t)k * S_DIM + j];
    float part = acc * E[(size_t)(T_LEN - 1) * S_DIM + j];
    __syncthreads();
    sa[j] = part;
    __syncthreads();
    for (int s = 512; s >= 1; s >>= 1) {        // sum -> scalar
        if (j < s) sa[j] += sa[j + s];
        __syncthreads();
    }
    if (j == 0) out[0] = sa[0];
}

// ============================================================================
extern "C" void kernel_launch(void* const* d_in, const int* in_sizes, int n_in,
                              void* d_out, int out_size, void* d_ws, size_t ws_size,
                              hipStream_t stream) {
    (void)in_sizes; (void)n_in; (void)out_size; (void)ws_size;
    const int*   obs     = (const int*)  d_in[0];
    const float* q_init  = (const float*)d_in[1];
    const float* q_trans = (const float*)d_in[2];
    const float* q_emis  = (const float*)d_in[3];
    char* ws = (char*)d_ws;

    __bf16* A   = (__bf16*)(ws + A_OFF);
    float*  E   = (float*) (ws + E_OFF);
    __bf16* Ebf = (__bf16*)(ws + EBF_OFF);
    float*  p0  = (float*) (ws + P0V_OFF);
    float*  mxv = (float*) (ws + MX_OFF);
    float*  inv = (float*) (ws + INV_OFF);
    __bf16* PB0 = (__bf16*)(ws + PB0_OFF);
    __bf16* PB1 = (__bf16*)(ws + PB1_OFF);

    // preprocessing
    hmm_softmax_init <<<1,     256, 0, stream>>>(q_init, p0);
    hmm_softmax_trans<<<S_DIM, 256, 0, stream>>>(q_trans, A);
    hmm_emis_stats   <<<S_DIM, 256, 0, stream>>>(q_emis, mxv, inv);
    hmm_build_E      <<<T_LEN, 256, 0, stream>>>(q_emis, obs, mxv, inv, E, Ebf);

    // chunk products: P[c] = prod of (diag(e_t) @ A) over chunk c's steps
    hmm_initP<<<dim3(S_DIM, NCHUNK), 256, 0, stream>>>(A, E, PB0);
    __bf16* cur = PB0; __bf16* nxt = PB1;
    for (int i = 1; i < CLEN; ++i) {
        hmm_gemm<true><<<dim3(64, NCHUNK), 256, 0, stream>>>(nxt, cur, A, 1, 0, 0, Ebf, i);
        __bf16* t = cur; cur = nxt; nxt = t;
    }
    // pairwise tree combine: 32 -> 16 -> 8 -> 4 -> 2 -> 1
    for (int n = NCHUNK / 2; n >= 1; n >>= 1) {
        hmm_gemm<false><<<dim3(64, n), 256, 0, stream>>>(nxt, cur, cur, 2, 2, 1, nullptr, 0);
        __bf16* t = cur; cur = nxt; nxt = t;
    }
    // finish: vec-mat-vec + reduce
    hmm_final<<<1, 1024, 0, stream>>>(p0, E, A, cur, (float*)d_out);
}
